// FullCliffordConv2d_16131897164423
// MI455X (gfx1250) — matrix operand order repaired and checked by reading, not compiled, not run on hardware
//
#include <hip/hip_runtime.h>

// ---------------------------------------------------------------------------
// FullCliffordConv2d on gfx1250 via V_WMMA_F32_16X16X32_BF16 with wide async
// staging (GLOBAL_LOAD_ASYNC_TO_LDS_B128 + S_WAIT_ASYNCCNT).
//   x:      [8][256][128][128] f32, channel c = i*16 + m  (i=in-block, m=mv)
//   weight: [16][16][3][3][16] f32  (o, i, kh, kw, m)
//   bias:   [16][16] f32            (o, m)
//   out:    [8][256][128][128] f32, channel c = o*16 + m
// Per (b, m): a 16->16 channel 3x3 conv == WMMA tiles
//   D[o][pixel] += sum_K A[o][K] * B[K][pixel],  K = (tap, in-channel) pairs
// ---------------------------------------------------------------------------

typedef __attribute__((ext_vector_type(16))) __bf16 v16bf;
typedef __attribute__((ext_vector_type(2)))  __bf16 v2bf;
typedef __attribute__((ext_vector_type(8)))  float  v8f;

#define BATCH 8
#define IB 16      // in blocks
#define OB 16      // out blocks
#define MV 16      // multivector dim (= groups)
#define HH 128
#define WW 128
#define KS 3
#define ROWS 4            // output rows per wave-task
#define SROWS (ROWS + 2)  // staged input rows (halo)
#define SPX 18            // compute-tile pixels (16 + 2 halo)
#define SPX_PAD 20        // padded pixel stride in bf16 LDS tile
#define SPXM 24           // staged memory pixels per row = 6 aligned b128 chunks
#define CHUNKS 6          // b128 chunks per (row, channel)
#define WAVES 4           // waves per block (wave32)

__global__ __launch_bounds__(WAVES * 32)
void clifford_conv_wmma_kernel(const float* __restrict__ x,
                               const float* __restrict__ wgt,
                               const float* __restrict__ bias,
                               float* __restrict__ out) {
    // raw f32 staging target for async b128 loads: [row][channel][24 px]
    __shared__ __align__(16) float  s_raw[WAVES][SROWS][IB][SPXM];
    // bf16 compute tile, channel-innermost so a B-fragment (16 channels at one
    // (row,pixel)) is one contiguous 32B-aligned LDS read per lane.
    __shared__ __align__(32) __bf16 s_x[WAVES][SROWS][SPX_PAD][IB];

    const int lane = threadIdx.x & 31;
    const int wave = threadIdx.x >> 5;

    // task = (b, m, h-block, w-tile); 8*16*32*8 = 32768 tasks, 4 per block
    int t = blockIdx.x * WAVES + wave;
    const int wt = t & 7;   t >>= 3;   // w tile: 0..7
    const int hb = t & 31;  t >>= 5;   // h block: 0..31
    const int m  = t & 15;  t >>= 4;   // mv component
    const int b  = t;                  // batch
    const int w0 = wt * 16;
    const int h0 = hb * ROWS;

    const bool hi  = lane >= 16;       // upper half-wave
    const int  l15 = lane & 15;

    const float* xb = x + ((size_t)b * (IB * MV) + m) * (HH * WW);
    const float* wbase = wgt + m;      // innermost stride of weight is MV

    __builtin_prefetch(wbase, 0, 3);   // weights are tiny & hot -> global_prefetch

    // ---- stage rows h0-1..h0+4, pixels w0-4..w0+19 (6 x 16B chunks, each
    // fully in- or out-of-bounds since w0 % 16 == 0), all 16 channels.
    // In-bounds chunks: GLOBAL_LOAD_ASYNC_TO_LDS_B128 (ASYNCcnt, no VGPR trip).
    // OOB chunks: float4 zero via ds store.  Generic shared-pointer low 32
    // bits == LDS byte address.
    for (int c = lane; c < SROWS * IB * CHUNKS; c += 32) {   // 18 iterations
        const int ck = c % CHUNKS;
        const int ci = (c / CHUNKS) & 15;
        const int rr = c / (CHUNKS * IB);
        const int hrow = h0 + rr - 1;
        const int wcol = w0 - 4 + ck * 4;    // 16B-aligned chunk start
        if (hrow >= 0 && hrow < HH && wcol >= 0 && wcol < WW) {
            unsigned int lds_addr =
                (unsigned int)(unsigned long long)&s_raw[wave][rr][ci][ck * 4];
            unsigned long long gaddr = (unsigned long long)
                (xb + (size_t)ci * MV * (HH * WW) + hrow * WW + wcol);
            asm volatile("global_load_async_to_lds_b128 %0, %1, off"
                         :: "v"(lds_addr), "v"(gaddr) : "memory");
        } else {
            *(float4*)&s_raw[wave][rr][ci][ck * 4] = make_float4(0.f, 0.f, 0.f, 0.f);
        }
    }
    asm volatile("s_wait_asynccnt 0x0" ::: "memory");
    __syncthreads();

    // ---- convert pass: LDS f32 -> LDS bf16 tile (channel-innermost), packing
    // two adjacent channels per dword store (v_cvt_pk_bf16_f32 friendly).
    // Tile pixel p corresponds to staged column p + 3 (w0-1 .. w0+16).
    for (int e = lane; e < SROWS * (IB / 2) * SPX; e += 32) {  // 27 iterations
        const int pxt = e % SPX;
        const int cp  = (e / SPX) & 7;
        const int rr  = e / (SPX * (IB / 2));
        v2bf v;
        v[0] = (__bf16)s_raw[wave][rr][2 * cp][pxt + 3];
        v[1] = (__bf16)s_raw[wave][rr][2 * cp + 1][pxt + 3];
        *(v2bf*)&s_x[wave][rr][pxt][2 * cp] = v;
    }
    __syncthreads();

    // ---- A fragments (weights), 5 fragments of K=32 (= 2 taps x 16 channels).
    // 16-bit A 16x32 layout (contiguous, matching B): lane<16 -> M=lane, elems 0-15 = K0-15
    //                        lane>=16 -> M=lane-16, elems 0-15 = K16-31
    // K ordering: K<16 -> tap 2p, ch=K ; K>=16 -> tap 2p+1, ch=K-16
    const int o = l15;
    v16bf afrag[5];
#pragma unroll
    for (int p = 0; p < 5; ++p) {
        v16bf a;
#pragma unroll
        for (int j = 0; j < 16; ++j) {
            const int tap = 2 * p + (hi ? 1 : 0);
            const int ci  = j;
            float v = 0.0f;
            if (tap < 9)  // tap 9 is K-padding -> zero
                v = wbase[(size_t)(((o * IB + ci) * KS + tap / 3) * KS + tap % 3) * MV];
            a[j] = (__bf16)v;
        }
        afrag[p] = a;
    }

    // ---- bias into C layout: lane<16 -> elem e = row M=e; lane>=16 -> M=e+8
    v8f cbias;
#pragma unroll
    for (int e = 0; e < 8; ++e)
        cbias[e] = bias[(e + (hi ? 8 : 0)) * MV + m];

    float* ob = out + ((size_t)b * (OB * MV) + m) * (HH * WW);

#pragma unroll
    for (int r = 0; r < ROWS; ++r) {
        v8f acc = cbias;
#pragma unroll
        for (int p = 0; p < 5; ++p) {
            // B 32x16 bf16 layout: lane<16 holds K=0..15 (tap 2p) at col N=lane,
            // lane>=16 holds K=16..31 (tap 2p+1) at col N=lane-16.
            int tap = 2 * p + (hi ? 1 : 0);
            if (tap > 8) tap = 8;  // padded half: A is zero there, keep B finite
            const int dh = tap / 3, dw = tap % 3;
            const v16bf bfrag = *(const v16bf*)(&s_x[wave][r + dh][l15 + dw][0]);
            acc = __builtin_amdgcn_wmma_f32_16x16x32_bf16(
                false, afrag[p], false, bfrag, (short)0, acc, false, false);
        }
        const int hrow = h0 + r;
#pragma unroll
        for (int e = 0; e < 8; ++e) {
            const int oo = e + (hi ? 8 : 0);
            ob[(size_t)oo * MV * (HH * WW) + hrow * WW + w0 + l15] = acc[e];
        }
    }
}

extern "C" void kernel_launch(void* const* d_in, const int* in_sizes, int n_in,
                              void* d_out, int out_size, void* d_ws, size_t ws_size,
                              hipStream_t stream) {
    (void)in_sizes; (void)n_in; (void)out_size; (void)d_ws; (void)ws_size;
    const float* x    = (const float*)d_in[0];
    const float* wgt  = (const float*)d_in[1];
    const float* bias = (const float*)d_in[2];
    float* out = (float*)d_out;

    const int tasks  = BATCH * MV * (HH / ROWS) * (WW / 16);  // 32768
    const int blocks = tasks / WAVES;                         // 8192
    hipLaunchKernelGGL(clifford_conv_wmma_kernel,
                       dim3(blocks), dim3(WAVES * 32), 0, stream,
                       x, wgt, bias, out);
}
